// GCNBase_9448928051675
// MI455X (gfx1250) — matrix operand order repaired and checked by reading, not compiled, not run on hardware
//
#include <hip/hip_runtime.h>

// ---------------------------------------------------------------------------
// 2-layer GCN for MI455X (gfx1250, wave32).
// GEMMs via v_wmma_f32_16x16x32_f16 (f16 in / f32 acc) with weights pre-packed
// into WMMA B-fragment layout (two b128 loads per fragment per lane);
// aggregation via warp-per-edge global_atomic_add_f32 (gathered matrices fit
// in the 192MB L2).
// ---------------------------------------------------------------------------

#define N_NODES 100000
#define N_EDGES 1600000
#define F_IN    128
#define HID     64
#define C_OUT   40

typedef __attribute__((ext_vector_type(16))) _Float16 v16h;
typedef __attribute__((ext_vector_type(8)))  _Float16 v8h;
typedef __attribute__((ext_vector_type(8)))  float    v8f;
typedef __attribute__((ext_vector_type(4)))  float    v4f;

// ---------------- degree / normalization ----------------

__global__ __launch_bounds__(256) void k_deg_init(float* __restrict__ deg) {
  int i = blockIdx.x * 256 + threadIdx.x;
  if (i < N_NODES) deg[i] = 1.0f;  // self-loop contributes 1
}

__global__ __launch_bounds__(256) void k_deg_edges(const int* __restrict__ dst,
                                                   float* __restrict__ deg) {
  int e = blockIdx.x * 256 + threadIdx.x;
  if (e < N_EDGES) atomicAdd(&deg[dst[e]], 1.0f);
}

__global__ __launch_bounds__(256) void k_dinv(float* __restrict__ deg) {
  int i = blockIdx.x * 256 + threadIdx.x;
  if (i < N_NODES) deg[i] = rsqrtf(deg[i]);  // deg >= 1 always (self-loop)
}

// ---------------- weight pre-pack into WMMA B-fragment layout --------------
// B-fragment layout (16-bit B, 32x16 KxN): lane L holds column N = L&15,
// K-half 8*(L>>4); element j -> K = kk*32 + 8*(L>>4) + (j&7) + 16*(j>>3).
// Packed buffer: frag fi = nt*KSTEPS + kk; f16 value at
//   Wpack[(fi*32 + lane)*16 + j]
// so each lane's 16 f16 (32B) are contiguous -> two coalesced b128 loads.

// W1: 128x64 -> 4 nt x 4 kk x 32 lanes x 16 = 8192 f16
__global__ __launch_bounds__(256) void k_pack_w1(const float* __restrict__ W,
                                                 _Float16* __restrict__ P) {
  int t = blockIdx.x * 256 + threadIdx.x;
  if (t >= 4 * 4 * 32 * 16) return;
  int j    = t & 15;
  int lane = (t >> 4) & 31;
  int kk   = (t >> 9) & 3;
  int nt   = t >> 11;
  int k    = kk * 32 + 8 * (lane >> 4) + (j & 7) + ((j >> 3) << 4);
  int col  = nt * 16 + (lane & 15);
  P[t] = (_Float16)W[k * HID + col];
}

// W2: 64x40 (padded to 48 cols) -> 3 nt x 2 kk x 32 lanes x 16 = 3072 f16
__global__ __launch_bounds__(256) void k_pack_w2(const float* __restrict__ W,
                                                 _Float16* __restrict__ P) {
  int t = blockIdx.x * 256 + threadIdx.x;
  if (t >= 3 * 2 * 32 * 16) return;
  int j    = t & 15;
  int lane = (t >> 4) & 31;
  int kk   = (t >> 8) & 1;
  int nt   = t >> 9;
  int k    = kk * 32 + 8 * (lane >> 4) + (j & 7) + ((j >> 3) << 4);
  int col  = nt * 16 + (lane & 15);
  P[t] = (col < C_OUT) ? (_Float16)W[k * C_OUT + col] : (_Float16)0.0f;
}

__device__ inline v16h load_frag(const _Float16* __restrict__ P, int fi, int lane) {
  const _Float16* base = P + ((size_t)(fi * 32 + lane) << 4);  // 32B/lane, aligned
  v8h lo = *(const v8h*)(base);
  v8h up = *(const v8h*)(base + 8);
  v16h b;
#pragma unroll
  for (int j = 0; j < 8; ++j) { b[j] = lo[j]; b[j + 8] = up[j]; }
  return b;
}

// ---------------- GEMM1: XW1[100000x64] = X[100000x128] @ W1[128x64] -------
// Wave-per-16-row-strip (100000 = 6250*16); 4 K-steps x 4 N-tiles = 16 WMMAs.
// ISA 16-bit A layout: lane&15 = M, lane>>4 picks K-half (8);
//   elems 0..7 -> K = khalf+0..7, elems 8..15 -> K = khalf+16..23.
// C/D: vgpr r, lane L -> M = 8*(L>>4)+r, N = L&15.

__global__ __launch_bounds__(128) void k_gemm1(const float* __restrict__ X,
                                               const _Float16* __restrict__ P,
                                               float* __restrict__ Y) {
  const int lane  = threadIdx.x & 31;
  const int wave  = threadIdx.x >> 5;
  const int strip = blockIdx.x * 4 + wave;
  if (strip >= N_NODES / 16) return;       // wave-uniform exit (EXEC stays all-1)
  const int row0  = strip * 16;
  const int rlo   = lane & 15;
  const int hi    = lane >> 4;
  const int arow  = row0 + rlo;
  const int khalf = hi * 8;

  v16h a[4];
#pragma unroll
  for (int kk = 0; kk < 4; ++kk) {
    const float* base = X + (size_t)arow * F_IN + kk * 32 + khalf;  // 32B aligned
    v4f f0 = *(const v4f*)(base);
    v4f f1 = *(const v4f*)(base + 4);
    v4f f2 = *(const v4f*)(base + 16);
    v4f f3 = *(const v4f*)(base + 20);
#pragma unroll
    for (int j = 0; j < 4; ++j) {
      a[kk][j]      = (_Float16)f0[j];
      a[kk][j + 4]  = (_Float16)f1[j];
      a[kk][j + 8]  = (_Float16)f2[j];
      a[kk][j + 12] = (_Float16)f3[j];
    }
  }

#pragma unroll
  for (int nt = 0; nt < 4; ++nt) {
    const int col = nt * 16 + rlo;
    v8f acc = {};
#pragma unroll
    for (int kk = 0; kk < 4; ++kk) {
      v16h b = load_frag(P, nt * 4 + kk, lane);
      acc = __builtin_amdgcn_wmma_f32_16x16x32_f16(false, a[kk], false, b,
                                                   (short)0, acc, false, false);
    }
#pragma unroll
    for (int r = 0; r < 8; ++r) {
      int m = hi * 8 + r;
      Y[(size_t)(row0 + m) * HID + col] = acc[r];
    }
  }
}

// ---------------- layer-1 aggregation ----------------

// out1 = xw1 * dinv^2 + b1   (self-loop folded into init; bias added once)
__global__ __launch_bounds__(256) void k_init1(const float* __restrict__ xw1,
                                               const float* __restrict__ dinv,
                                               const float* __restrict__ b1,
                                               float* __restrict__ out1) {
  int idx = blockIdx.x * 256 + threadIdx.x;
  if (idx < N_NODES * HID) {
    int i = idx >> 6;
    int f = idx & 63;
    float di = dinv[i];
    out1[idx] = xw1[idx] * di * di + b1[f];
  }
}

// warp per edge: lane handles features {lane, lane+32}
__global__ __launch_bounds__(256) void k_edge1(const int* __restrict__ src,
                                               const int* __restrict__ dst,
                                               const float* __restrict__ dinv,
                                               const float* __restrict__ xw1,
                                               float* __restrict__ out1) {
  int gw = (blockIdx.x * 256 + threadIdx.x) >> 5;  // warp-uniform edge id
  if (gw >= N_EDGES) return;
  int lane = threadIdx.x & 31;
  int s = src[gw];
  int d = dst[gw];
  float w = dinv[s] * dinv[d];
  const float* xs = xw1 + (size_t)s * HID;
  float*       od = out1 + (size_t)d * HID;
  atomicAdd(&od[lane],      xs[lane]      * w);
  atomicAdd(&od[lane + 32], xs[lane + 32] * w);
}

// h = f16(relu(out1))
__global__ __launch_bounds__(256) void k_relu_h(const float* __restrict__ out1,
                                                _Float16* __restrict__ h) {
  int idx = blockIdx.x * 256 + threadIdx.x;
  if (idx < N_NODES * HID) h[idx] = (_Float16)fmaxf(out1[idx], 0.0f);
}

// ---------------- GEMM2: XW2[100000x40] = H[100000x64] @ W2[64x40] ---------
// K=64 -> 2 K-steps; N padded 40 -> 48 (3 tiles; pad handled in packed W,
// stores guarded per-lane).

__global__ __launch_bounds__(128) void k_gemm2(const _Float16* __restrict__ H,
                                               const _Float16* __restrict__ P,
                                               float* __restrict__ Y) {
  const int lane  = threadIdx.x & 31;
  const int wave  = threadIdx.x >> 5;
  const int strip = blockIdx.x * 4 + wave;
  if (strip >= N_NODES / 16) return;
  const int row0  = strip * 16;
  const int rlo   = lane & 15;
  const int hi    = lane >> 4;
  const int arow  = row0 + rlo;
  const int khalf = hi * 8;

  v16h a[2];
#pragma unroll
  for (int kk = 0; kk < 2; ++kk) {
    const _Float16* base = H + (size_t)arow * HID + kk * 32 + khalf;  // 16B aligned
    v8h lo = *(const v8h*)(base);
    v8h up = *(const v8h*)(base + 16);
#pragma unroll
    for (int j = 0; j < 8; ++j) {
      a[kk][j]     = lo[j];
      a[kk][j + 8] = up[j];
    }
  }

#pragma unroll
  for (int nt = 0; nt < 3; ++nt) {
    const int col = nt * 16 + rlo;
    v8f acc = {};
#pragma unroll
    for (int kk = 0; kk < 2; ++kk) {
      v16h b = load_frag(P, nt * 2 + kk, lane);
      acc = __builtin_amdgcn_wmma_f32_16x16x32_f16(false, a[kk], false, b,
                                                   (short)0, acc, false, false);
    }
    if (col < C_OUT) {
#pragma unroll
      for (int r = 0; r < 8; ++r) {
        int m = hi * 8 + r;
        Y[(size_t)(row0 + m) * C_OUT + col] = acc[r];
      }
    }
  }
}

// ---------------- layer-2 aggregation ----------------

__global__ __launch_bounds__(256) void k_init2(const float* __restrict__ xw2,
                                               const float* __restrict__ dinv,
                                               const float* __restrict__ b2,
                                               float* __restrict__ out) {
  int idx = blockIdx.x * 256 + threadIdx.x;
  if (idx < N_NODES * C_OUT) {
    int i = idx / C_OUT;
    int f = idx - i * C_OUT;
    float di = dinv[i];
    out[idx] = xw2[idx] * di * di + b2[f];
  }
}

__global__ __launch_bounds__(256) void k_edge2(const int* __restrict__ src,
                                               const int* __restrict__ dst,
                                               const float* __restrict__ dinv,
                                               const float* __restrict__ xw2,
                                               float* __restrict__ out) {
  int gw = (blockIdx.x * 256 + threadIdx.x) >> 5;
  if (gw >= N_EDGES) return;
  int lane = threadIdx.x & 31;
  int s = src[gw];
  int d = dst[gw];
  float w = dinv[s] * dinv[d];
  const float* xs = xw2 + (size_t)s * C_OUT;
  float*       od = out + (size_t)d * C_OUT;
  atomicAdd(&od[lane], xs[lane] * w);            // lanes 0..31 all < 40
  if (lane < C_OUT - 32)
    atomicAdd(&od[lane + 32], xs[lane + 32] * w);
}

// ---------------- launch ----------------

extern "C" void kernel_launch(void* const* d_in, const int* in_sizes, int n_in,
                              void* d_out, int out_size, void* d_ws, size_t ws_size,
                              hipStream_t stream) {
  const float* x   = (const float*)d_in[0];
  const int*   ei  = (const int*)d_in[1];   // [2, E] flat: src then dst
  const float* W1  = (const float*)d_in[2];
  const float* b1  = (const float*)d_in[3];
  const float* W2  = (const float*)d_in[4];
  const float* b2  = (const float*)d_in[5];
  float*       out = (float*)d_out;

  const int* src = ei;
  const int* dst = ei + N_EDGES;

  // Workspace layout (aliased; total ~51.8 MB):
  //   [0, 512KB)     : deg -> dinv                 (N floats)
  //   bufA (25.6 MB) : XW1 f32, later H f16
  //   bufB (25.6 MB) : OUT1 f32, later XW2 f32
  //   Wpack1 (16 KB) : packed f16 W1 fragments
  //   Wpack2 (6 KB)  : packed f16 W2 fragments (N-padded)
  char* ws = (char*)d_ws;
  size_t off = 0;
  float*    dinv = (float*)(ws + off);  off += (size_t)512 * 1024;
  float*    bufA = (float*)(ws + off);  off += (size_t)N_NODES * HID * 4;
  float*    bufB = (float*)(ws + off);  off += (size_t)N_NODES * HID * 4;
  _Float16* wp1  = (_Float16*)(ws + off); off += (size_t)4 * 4 * 32 * 16 * 2;
  _Float16* wp2  = (_Float16*)(ws + off);

  float*    xw1  = bufA;
  float*    out1 = bufB;
  _Float16* h    = (_Float16*)bufA;   // reuses XW1 after edge pass 1
  float*    xw2  = bufB;              // reuses OUT1 after relu

  const int strips     = N_NODES / 16;             // 6250 exactly
  const int gemm_grid  = (strips + 3) / 4;         // 4 waves per 128-thr block
  const int node_grid  = (N_NODES + 255) / 256;
  const int edge_grid  = (N_EDGES + 255) / 256;
  const int ewarp_grid = (N_EDGES * 32 + 255) / 256;   // warp per edge
  const int nh_grid    = (N_NODES * HID + 255) / 256;
  const int nc_grid    = (N_NODES * C_OUT + 255) / 256;

  // weight pre-pack + normalization (independent, tiny)
  k_pack_w1  <<<(4 * 4 * 32 * 16 + 255) / 256, 256, 0, stream>>>(W1, wp1);
  k_pack_w2  <<<(3 * 2 * 32 * 16 + 255) / 256, 256, 0, stream>>>(W2, wp2);
  k_deg_init <<<node_grid, 256, 0, stream>>>(dinv);
  k_deg_edges<<<edge_grid, 256, 0, stream>>>(dst, dinv);
  k_dinv     <<<node_grid, 256, 0, stream>>>(dinv);

  // layer 1
  k_gemm1 <<<gemm_grid, 128, 0, stream>>>(x, wp1, xw1);
  k_init1 <<<nh_grid, 256, 0, stream>>>(xw1, dinv, b1, out1);
  k_edge1 <<<ewarp_grid, 256, 0, stream>>>(src, dst, dinv, xw1, out1);
  k_relu_h<<<nh_grid, 256, 0, stream>>>(out1, h);

  // layer 2
  k_gemm2 <<<gemm_grid, 128, 0, stream>>>(h, wp2, xw2);
  k_init2 <<<nc_grid, 256, 0, stream>>>(xw2, dinv, b2, out);
  k_edge2 <<<ewarp_grid, 256, 0, stream>>>(src, dst, dinv, xw2, out);
}
